// MLPs_78314433675304
// MI455X (gfx1250) — compile-verified, matrix operand order checked by reading, not run on hardware
//
#include <hip/hip_runtime.h>
#include <hip/hip_bf16.h>

typedef __attribute__((ext_vector_type(16))) _Float16 v16h;
typedef __attribute__((ext_vector_type(8)))  float    v8f;

#define NROWS 65536   // B * N_OPE
#define NOPE  8192
#define NMA   128
#define NBUF  64

// ---------------------------------------------------------------------------
// Weight convert: fp32 [K x N] -> fp16 [padK x N], zero-padding rows K..padK-1
// ---------------------------------------------------------------------------
__global__ void cvt_pad_kernel(const float* __restrict__ src, _Float16* __restrict__ dst,
                               int K, int N, int padK) {
    int i = blockIdx.x * blockDim.x + threadIdx.x;
    int total = padK * N;
    if (i >= total) return;
    int k = i / N;
    dst[i] = (k < K) ? (_Float16)src[i] : (_Float16)0.0f;
}

// ---------------------------------------------------------------------------
// Aggregation: build X[rel] = agg features, f16, padded K=16 -> 32 with zeros.
// One thread per row (g = b*8192 + n), 16-wide f32 accumulate in registers.
// ---------------------------------------------------------------------------
__global__ void agg_kernel(int rel,
                           const int* __restrict__ adjA,
                           const int* __restrict__ adjB,
                           const float* __restrict__ fOpes,
                           const float* __restrict__ fMas,
                           const float* __restrict__ fBuf,
                           _Float16* __restrict__ X) {
    int g = blockIdx.x * blockDim.x + threadIdx.x;
    if (g >= NROWS) return;
    int b = g >> 13;          // / 8192
    int n = g & (NOPE - 1);

    float acc[16];
#pragma unroll
    for (int d = 0; d < 16; ++d) acc[d] = 0.0f;

    if (rel == 0) {                                   // h_pre: f[b, n+1]
        if (n < NOPE - 1) {
            const float* s = fOpes + (size_t)(g + 1) * 16;
#pragma unroll
            for (int d = 0; d < 16; ++d) acc[d] = s[d];
        }
    } else if (rel == 1) {                            // h_sub: f[b, n-1]
        if (n > 0) {
            const float* s = fOpes + (size_t)(g - 1) * 16;
#pragma unroll
            for (int d = 0; d < 16; ++d) acc[d] = s[d];
        }
    } else if (rel == 5) {                            // identity
        const float* s = fOpes + (size_t)g * 16;
#pragma unroll
        for (int d = 0; d < 16; ++d) acc[d] = s[d];
    } else if (rel == 2 || rel == 3) {                // adj @ feat_mas
        const int* arow = adjA + (size_t)n * NMA;
        for (int k = 0; k < NMA; ++k) {
            int a = arow[k];
            if (a) {
                float w = (float)a;
                const float* s = fMas + ((size_t)b * NMA + k) * 16;
#pragma unroll
                for (int d = 0; d < 16; ++d) acc[d] += w * s[d];
            }
        }
    } else {                                          // rel 4: (buf|buf_out) @ feat_buf
        const int* arow = adjA + (size_t)n * NBUF;
        const int* brow = adjB + (size_t)n * NBUF;
        for (int k = 0; k < NBUF; ++k) {
            if (arow[k] > 0 || brow[k] > 0) {
                const float* s = fBuf + ((size_t)b * NBUF + k) * 16;
#pragma unroll
                for (int d = 0; d < 16; ++d) acc[d] += s[d];
            }
        }
    }

    _Float16* xo = X + (size_t)g * 32;
#pragma unroll
    for (int d = 0; d < 16; ++d) {
        xo[d]      = (_Float16)acc[d];
        xo[16 + d] = (_Float16)0.0f;
    }
}

// ---------------------------------------------------------------------------
// WMMA GEMM: C[M x NCOLS] = act(A[M x K](f16, lda) @ W[K x NCOLS](f16) + bias)
// Block = 256 threads = 8 waves; block tile 256(M) x TN(N); K step 32.
// Each wave owns a 32 x TN C slab: 2 M-fragments x NFRAG N-fragments
// -> 2*NFRAG WMMAs per K step; B fragments amortized over both M halves.
// NCOLS is compile-time so all column loops/branches are static and the
// v8f accumulators stay pinned (in-place v_wmma, no accumulator movs).
// ---------------------------------------------------------------------------
#define SA_STRIDE 34   // halfs; odd-dword padded to dodge LDS bank conflicts
#define SB_STRIDE 34

template <int NCOLS, bool ELU, bool OUT_F16>
__global__ __launch_bounds__(256)
void wmma_mlp_gemm(const _Float16* __restrict__ A, int lda,
                   const _Float16* __restrict__ Bw,
                   const float* __restrict__ bias,
                   void* __restrict__ Cout, int ldc,
                   int K) {
    constexpr int TN    = (NCOLS >= 64) ? 64 : NCOLS;   // block tile width
    constexpr int NFRAG = TN / 16;                      // WMMA col fragments/wave

    __shared__ _Float16 sA[256 * SA_STRIDE];   // A tile, row-major [256][32]
    __shared__ _Float16 sBT[TN * SB_STRIDE];   // B tile, transposed [n][k]

    const int tid  = threadIdx.x;
    const int lane = tid & 31;
    const int wave = tid >> 5;
    const int h    = lane >> 4;     // half-wave select
    const int ln   = lane & 15;
    const int rowBase = blockIdx.x * 256;
    const int nBase   = blockIdx.y * TN;

    v8f c[2][NFRAG];
#pragma unroll
    for (int mi = 0; mi < 2; ++mi)
#pragma unroll
        for (int cf = 0; cf < NFRAG; ++cf)
#pragma unroll
            for (int v = 0; v < 8; ++v) c[mi][cf][v] = 0.0f;

    for (int k0 = 0; k0 < K; k0 += 32) {
        // --- stage A tile: 256 rows x 32 halfs, uint4 (8-half) vector loads
#pragma unroll
        for (int it = 0; it < 4; ++it) {
            int t = tid + it * 256;            // 0..1023
            int row = t >> 2;
            int chunk = t & 3;
            const uint4* src =
                (const uint4*)(A + (size_t)(rowBase + row) * lda + k0 + chunk * 8);
            uint4 v = *src;
            unsigned int* d = (unsigned int*)(sA + row * SA_STRIDE + chunk * 8);
            d[0] = v.x; d[1] = v.y; d[2] = v.z; d[3] = v.w;
        }
        // --- stage B tile transposed: sBT[n][k], n in [0,TN), k in [0,32)
#pragma unroll
        for (int it = 0; it < (32 * TN) / 256; ++it) {
            int t = tid + it * 256;
            int k = t / TN;
            int n = t % TN;
            sBT[n * SB_STRIDE + k] = Bw[(size_t)(k0 + k) * NCOLS + nBase + n];
        }
        __syncthreads();

        // --- A fragments: lane ln = row, element e -> K = 16*(e>=8) + 8*h + (e&7)
        union { v16h v; unsigned int u[8]; } af[2];
#pragma unroll
        for (int mi = 0; mi < 2; ++mi) {
            const int mrow = wave * 32 + mi * 16 + ln;
#pragma unroll
            for (int gg = 0; gg < 8; ++gg) {
                int kp = (gg < 4) ? (2 * gg + 8 * h) : (16 + 2 * (gg - 4) + 8 * h);
                af[mi].u[gg] = *(const unsigned int*)(sA + mrow * SA_STRIDE + kp);
            }
        }
        // --- NFRAG column fragments, each reused by both M halves
#pragma unroll
        for (int cf = 0; cf < NFRAG; ++cf) {
            union { v16h v; unsigned int u[8]; } bf;
            const int nn = cf * 16 + ln;
#pragma unroll
            for (int gg = 0; gg < 8; ++gg) {
                int kp = (gg < 4) ? (2 * gg + 8 * h) : (16 + 2 * (gg - 4) + 8 * h);
                bf.u[gg] = *(const unsigned int*)(sBT + nn * SB_STRIDE + kp);
            }
#pragma unroll
            for (int mi = 0; mi < 2; ++mi) {
                c[mi][cf] = __builtin_amdgcn_wmma_f32_16x16x32_f16(
                    false, af[mi].v, false, bf.v, (short)0, c[mi][cf], false, false);
            }
        }
        __syncthreads();
    }

    // --- epilogue: bias + optional ELU; C layout: row = v + 8*h, col = ln
#pragma unroll
    for (int mi = 0; mi < 2; ++mi) {
#pragma unroll
        for (int cf = 0; cf < NFRAG; ++cf) {
            int col = nBase + cf * 16 + ln;
            float bv = bias[col];
#pragma unroll
            for (int v = 0; v < 8; ++v) {
                int row = rowBase + wave * 32 + mi * 16 + v + 8 * h;
                float x = c[mi][cf][v] + bv;
                if (ELU) x = (x > 0.0f) ? x : (__expf(x) - 1.0f);
                if (OUT_F16)
                    ((_Float16*)Cout)[(size_t)row * ldc + col] = (_Float16)x;
                else
                    ((float*)Cout)[(size_t)row * ldc + col] = x;
            }
        }
    }
}

// ---------------------------------------------------------------------------
// Host-side orchestration
// ---------------------------------------------------------------------------
extern "C" void kernel_launch(void* const* d_in, const int* in_sizes, int n_in,
                              void* d_out, int out_size, void* d_ws, size_t ws_size,
                              hipStream_t stream) {
    const int*   ope_ma_adj      = (const int*)d_in[0];
    const int*   ope_ma_adj_out  = (const int*)d_in[1];
    const int*   ope_buf_adj     = (const int*)d_in[2];
    const int*   ope_buf_adj_out = (const int*)d_in[3];
    // d_in[4] = batch_idxes (arange(B)) -- dense batch, unused
    const float* feat_opes = (const float*)d_in[5];
    const float* feat_mas  = (const float*)d_in[6];
    const float* feat_buf  = (const float*)d_in[7];

    // params pytree flattened with sorted dict keys:
    // gnn[i]: {b1,b2,b3,w1,w2,w3}, then proj: {b1,b2,b3,w1,w2,w3}
    const float *gb1[6], *gb2[6], *gb3[6], *gw1[6], *gw2[6], *gw3[6];
    for (int i = 0; i < 6; ++i) {
        gb1[i] = (const float*)d_in[8 + 6 * i + 0];
        gb2[i] = (const float*)d_in[8 + 6 * i + 1];
        gb3[i] = (const float*)d_in[8 + 6 * i + 2];
        gw1[i] = (const float*)d_in[8 + 6 * i + 3];
        gw2[i] = (const float*)d_in[8 + 6 * i + 4];
        gw3[i] = (const float*)d_in[8 + 6 * i + 5];
    }
    const float* pb1 = (const float*)d_in[44];
    const float* pb2 = (const float*)d_in[45];
    const float* pb3 = (const float*)d_in[46];
    const float* pw1 = (const float*)d_in[47];
    const float* pw2 = (const float*)d_in[48];
    const float* pw3 = (const float*)d_in[49];

    // Workspace layout (all 16B-aligned offsets)
    char* ws = (char*)d_ws;
    _Float16* X  = (_Float16*)(ws + 0);                       // 65536 x 32  (4 MB)
    _Float16* H1 = (_Float16*)(ws + (size_t)4194304);         // 65536 x 256 (32 MB)
    _Float16* H2 = (_Float16*)(ws + (size_t)37748736);        // 65536 x 256 (32 MB)
    _Float16* E  = (_Float16*)(ws + (size_t)71303168);        // 65536 x 96  (12 MB)
    char* wp = ws + (size_t)83886080;                         // f16 weights
    _Float16* Wg1[6]; _Float16* Wg2[6]; _Float16* Wg3[6];
    size_t off = 0;
    for (int i = 0; i < 6; ++i) { Wg1[i] = (_Float16*)(wp + off); off += 32 * 256 * 2; }
    for (int i = 0; i < 6; ++i) { Wg2[i] = (_Float16*)(wp + off); off += 256 * 256 * 2; }
    for (int i = 0; i < 6; ++i) { Wg3[i] = (_Float16*)(wp + off); off += 256 * 16 * 2; }
    _Float16* Wp1 = (_Float16*)(wp + off); off += 96 * 256 * 2;
    _Float16* Wp2 = (_Float16*)(wp + off); off += 256 * 256 * 2;
    _Float16* Wp3 = (_Float16*)(wp + off); off += 256 * 16 * 2;

    // 1) convert + pad weights to f16 (tiny, L2-resident afterwards)
    auto cvt = [&](const float* s, _Float16* d, int K, int N, int padK) {
        int tot = padK * N;
        cvt_pad_kernel<<<(tot + 255) / 256, 256, 0, stream>>>(s, d, K, N, padK);
    };
    for (int i = 0; i < 6; ++i) {
        cvt(gw1[i], Wg1[i], 16, 256, 32);
        cvt(gw2[i], Wg2[i], 256, 256, 256);
        cvt(gw3[i], Wg3[i], 256, 16, 256);
    }
    cvt(pw1, Wp1, 96, 256, 96);
    cvt(pw2, Wp2, 256, 256, 256);
    cvt(pw3, Wp3, 256, 16, 256);

    const dim3 blk(256);
    const dim3 gWide(NROWS / 256, 4);   // N=256, TN=64
    const dim3 gNarrow(NROWS / 256, 1); // N=16,  TN=16
    const dim3 gAgg(NROWS / 256);

    // 2) six relation MLPs -> concat buffer E
    for (int r = 0; r < 6; ++r) {
        const int* aA = (r == 2) ? ope_ma_adj
                      : (r == 3) ? ope_ma_adj_out
                                 : ope_buf_adj;
        agg_kernel<<<gAgg, blk, 0, stream>>>(r, aA, ope_buf_adj_out,
                                             feat_opes, feat_mas, feat_buf, X);
        wmma_mlp_gemm<256, true,  true ><<<gWide,   blk, 0, stream>>>(X,  32,  Wg1[r], gb1[r], H1, 256, 32);
        wmma_mlp_gemm<256, true,  true ><<<gWide,   blk, 0, stream>>>(H1, 256, Wg2[r], gb2[r], H2, 256, 256);
        wmma_mlp_gemm<16,  false, true ><<<gNarrow, blk, 0, stream>>>(H2, 256, Wg3[r], gb3[r], E + 16 * r, 96, 256);
    }

    // 3) projection MLP -> f32 output [B, N_OPE, 16]
    wmma_mlp_gemm<256, true,  true ><<<gWide,   blk, 0, stream>>>(E,  96,  Wp1, pb1, H1, 256, 96);
    wmma_mlp_gemm<256, true,  true ><<<gWide,   blk, 0, stream>>>(H1, 256, Wp2, pb2, H2, 256, 256);
    wmma_mlp_gemm<16,  false, false><<<gNarrow, blk, 0, stream>>>(H2, 256, Wp3, pb3, d_out, 16, 256);
}
